// ResGCNBlock_46497315946697
// MI455X (gfx1250) — matrix operand order repaired and checked by reading, not compiled, not run on hardware
//
#include <hip/hip_runtime.h>

// ---------------------------------------------------------------------------
// CDNA5 (gfx1250) wave32 WMMA types
// ---------------------------------------------------------------------------
typedef __attribute__((ext_vector_type(16))) __bf16 v16bf;
typedef __attribute__((ext_vector_type(8)))  float  v8f;

#define WAVE 32
#define COUT 128
#define CIN  256

static __device__ __forceinline__ unsigned short f2bf(float f) {
  union { float f; unsigned u; } v; v.f = f;
  unsigned u = v.u;
  u += 0x7FFFu + ((u >> 16) & 1u);          // round-to-nearest-even
  return (unsigned short)(u >> 16);
}

union BF16x16 { uint4 q[2]; unsigned short s[16]; v16bf v; };

// A-fragment (16x32 bf16, row-major source, lda = row stride in elements).
// ISA layout: lane&15 = M row; lane>>4 selects K-half; elems 0..7 -> K=kg*8..+7,
// elems 8..15 -> K=16+kg*8..+7  (two contiguous 16B chunks).
static __device__ __forceinline__ v16bf ld_a(const unsigned short* A, int lda,
                                             int m0, int k0, int lane, int M) {
  int m = m0 + (lane & 15); if (m >= M) m = M - 1;
  const unsigned short* p = A + (size_t)m * lda + k0 + ((lane >> 4) << 3);
  BF16x16 u;
  u.q[0] = *(const uint4*)p;
  u.q[1] = *(const uint4*)(p + 16);
  return u.v;
}

// A-fragment from a wave-private LDS tile (16 rows x 32 cols bf16, stride 32).
static __device__ __forceinline__ v16bf ld_a_lds(const unsigned short* P, int lane) {
  const unsigned short* p = P + (lane & 15) * 32 + ((lane >> 4) << 3);
  BF16x16 u;
  u.q[0] = *(const uint4*)p;
  u.q[1] = *(const uint4*)(p + 16);
  return u.v;
}

// B-fragment (32x16) from a TRANSPOSED matrix Bt (Ncols x K row-major):
// lane&15 = N col; lane>>4 = kg; elems 0..7 -> K=kg*8..+7, 8..15 -> K=16+kg*8..+7 -> 2x b128.
static __device__ __forceinline__ v16bf ld_bt(const unsigned short* Bt, int ldbt,
                                              int k0, int n0, int lane) {
  int n = n0 + (lane & 15);
  const unsigned short* p = Bt + (size_t)n * ldbt + k0 + ((lane >> 4) << 3);
  BF16x16 u;
  u.q[0] = *(const uint4*)p;
  u.q[1] = *(const uint4*)(p + 16);
  return u.v;
}

// B-fragment for K^T in attention: B(k,n) = K[key0+n][c0+k]; row-clamped.
static __device__ __forceinline__ v16bf ld_bT(const unsigned short* K, int key0,
                                              int c0, int lane, int Nq) {
  int key = key0 + (lane & 15); if (key >= Nq) key = Nq - 1;
  const unsigned short* p = K + (size_t)key * COUT + c0 + ((lane >> 4) << 3);
  BF16x16 u;
  u.q[0] = *(const uint4*)p;
  u.q[1] = *(const uint4*)(p + 16);
  return u.v;
}

static __device__ __forceinline__ v8f wmma_bf16(v16bf a, v16bf b, v8f c) {
  return __builtin_amdgcn_wmma_f32_16x16x32_bf16(false, a, false, b,
                                                 (short)0, c, false, false);
}

// ---------------------------------------------------------------------------
// Elementwise / graph kernels
// ---------------------------------------------------------------------------
__global__ void k_cvt_bf16(const float* __restrict__ in,
                           unsigned short* __restrict__ out, int n) {
  int i = blockIdx.x * blockDim.x + threadIdx.x;
  if (i < n) out[i] = f2bf(in[i]);
}

// transpose + convert: in[R x C] (row-major f32) -> out[C x R] (row-major bf16)
__global__ void k_cvt_bf16_T(const float* __restrict__ in,
                             unsigned short* __restrict__ out, int R, int C) {
  int i = blockIdx.x * blockDim.x + threadIdx.x;
  if (i < R * C) {
    int r = i / C, c = i - r * C;
    out[(size_t)c * R + r] = f2bf(in[i]);
  }
}

__global__ void k_zero_us(unsigned short* p, int n) {
  int i = blockIdx.x * blockDim.x + threadIdx.x;
  if (i < n) p[i] = 0;
}

__global__ void k_deg_init(float* deg, int n) {
  int i = blockIdx.x * blockDim.x + threadIdx.x;
  if (i < n) deg[i] = 1.0f;                       // self-loop
}

__global__ void k_deg_count(const int* __restrict__ dst, float* deg, int e) {
  int i = blockIdx.x * blockDim.x + threadIdx.x;
  if (i < e) atomicAdd(&deg[dst[i]], 1.0f);
}

__global__ void k_dis(const float* __restrict__ deg, float* dis, int n) {
  int i = blockIdx.x * blockDim.x + threadIdx.x;
  if (i < n) dis[i] = rsqrtf(deg[i]);
}

// agg[i] = h[i] * dis[i]^2   (self-loop contribution)
__global__ void k_selfloop(const float* __restrict__ h,
                           const float* __restrict__ dis,
                           float* __restrict__ agg, int n) {
  int i = blockIdx.x * blockDim.x + threadIdx.x;
  if (i < n * COUT) {
    int row = i >> 7;
    float d = dis[row];
    agg[i] = h[i] * d * d;
  }
}

// agg[dst] += h[src] * dis[src]*dis[dst]; 8 threads per edge, 16 floats each
__global__ void k_scatter(const int* __restrict__ src, const int* __restrict__ dst,
                          const float* __restrict__ h, const float* __restrict__ dis,
                          float* __restrict__ agg, int e) {
  int t = blockIdx.x * blockDim.x + threadIdx.x;
  int edge = t >> 3;
  if (edge >= e) return;
  int c0 = (t & 7) * 16;
  int s = src[edge], d = dst[edge];
  float nrm = dis[s] * dis[d];
  const float* hp = h + (size_t)s * COUT + c0;
  float*       ap = agg + (size_t)d * COUT + c0;
#pragma unroll
  for (int i = 0; i < 16; ++i) atomicAdd(ap + i, hp[i] * nrm);
}

// y = LN(x + bias) * g + beta, optional SiLU.  Wave-per-row (wave32, 4 elem/lane).
__global__ void k_bias_ln(const float* __restrict__ X, const float* __restrict__ bias,
                          const float* __restrict__ g, const float* __restrict__ beta,
                          float* __restrict__ Y, int M, int do_silu) {
  int lane = threadIdx.x & (WAVE - 1);
  int wv   = threadIdx.x >> 5;
  int row  = blockIdx.x * 8 + wv;
  if (row >= M) return;
  const float* xp = X + (size_t)row * COUT;
  float v[4], s = 0.f, s2 = 0.f;
#pragma unroll
  for (int i = 0; i < 4; ++i) {
    int c = lane * 4 + i;
    float t = xp[c] + bias[c];
    v[i] = t; s += t; s2 += t * t;
  }
#pragma unroll
  for (int off = 16; off > 0; off >>= 1) {
    s  += __shfl_xor(s,  off, WAVE);
    s2 += __shfl_xor(s2, off, WAVE);
  }
  float mu  = s * (1.0f / COUT);
  float var = s2 * (1.0f / COUT) - mu * mu;
  float inv = rsqrtf(var + 1e-5f);
  float* yp = Y + (size_t)row * COUT;
#pragma unroll
  for (int i = 0; i < 4; ++i) {
    int c = lane * 4 + i;
    float o = (v[i] - mu) * inv * g[c] + beta[c];
    if (do_silu) o = o / (1.0f + __expf(-o));
    yp[c] = o;
  }
}

// ---------------------------------------------------------------------------
// GEMM: C[MxCOUT] = A[MxK](bf16) * Bt^T (+bias), f32 accumulate.
// Bt is the transposed weight: COUT x K row-major, so B-frags are 2x b128.
// k-loop unrolled x2; all loads of a 64-deep k-slab are issued before the
// 16 WMMAs, so the second batch's loads overlap the first batch's math.
// 256 threads = 8 waves; block tile 128 rows; wave tile 16 rows x 128 cols.
// Requires K % 64 == 0 (K = 256 or 128 here).
// ---------------------------------------------------------------------------
__global__ void k_gemm_bf16(const unsigned short* __restrict__ A,
                            const unsigned short* __restrict__ Bt,
                            const float* __restrict__ bias,
                            float* __restrict__ C, int M, int K) {
  int lane = threadIdx.x & (WAVE - 1);
  int wv   = threadIdx.x >> 5;
  int row0 = blockIdx.x * 128 + wv * 16;
  v8f acc[8] = {};
  for (int k0 = 0; k0 < K; k0 += 64) {
    v16bf a0 = ld_a(A, K, row0, k0,      lane, M);
    v16bf a1 = ld_a(A, K, row0, k0 + 32, lane, M);
    v16bf b0[8], b1[8];
#pragma unroll
    for (int nt = 0; nt < 8; ++nt) b0[nt] = ld_bt(Bt, K, k0,      nt * 16, lane);
#pragma unroll
    for (int nt = 0; nt < 8; ++nt) b1[nt] = ld_bt(Bt, K, k0 + 32, nt * 16, lane);
#pragma unroll
    for (int nt = 0; nt < 8; ++nt) acc[nt] = wmma_bf16(a0, b0[nt], acc[nt]);
#pragma unroll
    for (int nt = 0; nt < 8; ++nt) acc[nt] = wmma_bf16(a1, b1[nt], acc[nt]);
  }
  int nlo = lane & 15, mh = lane >> 4;
#pragma unroll
  for (int nt = 0; nt < 8; ++nt) {
    int col = nt * 16 + nlo;
    float bv = bias ? bias[col] : 0.0f;
#pragma unroll
    for (int r = 0; r < 8; ++r) {
      int m = row0 + mh * 8 + r;
      if (m < M) C[(size_t)m * COUT + col] = acc[nt][r] + bv;
    }
  }
}

// ---------------------------------------------------------------------------
// Flash attention + residual + skip + SiLU epilogue.
// out = silu(softmax(QK^T/sqrt(128)) @ V + res + ident)
// Vt is V transposed (COUT x N) so P@V B-frags are contiguous b128 loads.
// 256 threads = 8 waves; wave owns 16 query rows; keys streamed 32 at a time.
// All fragment loads of a phase are issued before that phase's WMMAs.
// ---------------------------------------------------------------------------
__global__ void k_flash(const unsigned short* __restrict__ Qb,
                        const unsigned short* __restrict__ Kb,
                        const unsigned short* __restrict__ Vt,
                        const float* __restrict__ res,
                        const float* __restrict__ ident,
                        float* __restrict__ out, int Nq) {
  __shared__ float          Sbuf[8][16 * 32];
  __shared__ unsigned short Pbuf[8][16 * 32];
  __shared__ float          Arow[8][16];

  int lane = threadIdx.x & (WAVE - 1);
  int wv   = threadIdx.x >> 5;
  int q0   = blockIdx.x * 128 + wv * 16;
  int nlo  = lane & 15, mh = lane >> 4;
  const float scale = 0.08838834764831845f;       // 1/sqrt(128)

  v16bf qf[4];
#pragma unroll
  for (int c = 0; c < 4; ++c) qf[c] = ld_a(Qb, COUT, q0, c * 32, lane, Nq);

  v8f o[8] = {};
  float m_run = -1e30f, l_run = 0.0f;             // valid in lanes 0..15

  for (int key0 = 0; key0 < Nq; key0 += 32) {
    // ---- S = Q * K^T for 32 keys (two 16x16 tiles); batch loads first ----
    v16bf kt[8];
#pragma unroll
    for (int c = 0; c < 4; ++c) {
      kt[c]     = ld_bT(Kb, key0,      c * 32, lane, Nq);
      kt[4 + c] = ld_bT(Kb, key0 + 16, c * 32, lane, Nq);
    }
    v8f s0 = {}, s1 = {};
#pragma unroll
    for (int c = 0; c < 4; ++c) s0 = wmma_bf16(qf[c], kt[c],     s0);
#pragma unroll
    for (int c = 0; c < 4; ++c) s1 = wmma_bf16(qf[c], kt[4 + c], s1);
#pragma unroll
    for (int r = 0; r < 8; ++r) {                 // C layout -> LDS row-major
      Sbuf[wv][(mh * 8 + r) * 32 + nlo]      = s0[r] * scale;
      Sbuf[wv][(mh * 8 + r) * 32 + 16 + nlo] = s1[r] * scale;
    }
    __syncthreads();

    // ---- online softmax: lanes 0..15 each own one full query row ----
    if (lane < 16) {
      int row = lane;
      float mx = -1e30f;
      for (int j = 0; j < 32; ++j) {
        float val = (key0 + j < Nq) ? Sbuf[wv][row * 32 + j] : -1e30f;
        mx = fmaxf(mx, val);
      }
      float m_new = fmaxf(m_run, mx);
      float alpha = __expf(m_run - m_new);
      float psum  = 0.0f;
      for (int j = 0; j < 32; ++j) {
        float val = (key0 + j < Nq) ? Sbuf[wv][row * 32 + j] : -1e30f;
        float p = __expf(val - m_new);
        psum += p;
        Pbuf[wv][row * 32 + j] = f2bf(p);
      }
      l_run = l_run * alpha + psum;
      m_run = m_new;
      Arow[wv][row] = alpha;
    }
    __syncthreads();

    // ---- batch V-fragment loads, rescale O, then O += P * V ----
    v16bf vf[8];
#pragma unroll
    for (int nt = 0; nt < 8; ++nt)
      vf[nt] = ld_bt(Vt, Nq, key0, nt * 16, lane);     // contiguous b128 x2
#pragma unroll
    for (int r = 0; r < 8; ++r) {
      float al = Arow[wv][mh * 8 + r];
#pragma unroll
      for (int nt = 0; nt < 8; ++nt) o[nt][r] = o[nt][r] * al;
    }
    v16bf pf = ld_a_lds(&Pbuf[wv][0], lane);
#pragma unroll
    for (int nt = 0; nt < 8; ++nt) o[nt] = wmma_bf16(pf, vf[nt], o[nt]);
    __syncthreads();
  }

  if (lane < 16) Arow[wv][lane] = l_run;          // publish row sums
  __syncthreads();

#pragma unroll
  for (int r = 0; r < 8; ++r) {
    int row = q0 + mh * 8 + r;
    if (row >= Nq) continue;
    float linv = 1.0f / Arow[wv][mh * 8 + r];
#pragma unroll
    for (int nt = 0; nt < 8; ++nt) {
      int col = nt * 16 + nlo;
      float val = o[nt][r] * linv + res[(size_t)row * COUT + col]
                                  + ident[(size_t)row * COUT + col];
      out[(size_t)row * COUT + col] = val / (1.0f + __expf(-val));
    }
  }
}

// ---------------------------------------------------------------------------
// Host pipeline
// ---------------------------------------------------------------------------
extern "C" void kernel_launch(void* const* d_in, const int* in_sizes, int n_in,
                              void* d_out, int out_size, void* d_ws, size_t ws_size,
                              hipStream_t stream) {
  const float* x     = (const float*)d_in[0];
  const int*   ei    = (const int*)d_in[1];
  const float* W1    = (const float*)d_in[2];
  const float* b1    = (const float*)d_in[3];
  const float* W2    = (const float*)d_in[4];
  const float* b2    = (const float*)d_in[5];
  const float* ln1g  = (const float*)d_in[6];
  const float* ln1b  = (const float*)d_in[7];
  const float* ln2g  = (const float*)d_in[8];
  const float* ln2b  = (const float*)d_in[9];
  const float* Wq    = (const float*)d_in[10];
  const float* bq    = (const float*)d_in[11];
  const float* Wk    = (const float*)d_in[12];
  const float* bk    = (const float*)d_in[13];
  const float* Wv    = (const float*)d_in[14];
  const float* bv    = (const float*)d_in[15];
  const float* Ws    = (const float*)d_in[16];
  const float* bs    = (const float*)d_in[17];
  float* out = (float*)d_out;

  const int N = in_sizes[0] / CIN;
  const int E = in_sizes[1] / 2;
  const int* src = ei;
  const int* dst = ei + E;

  // ---- bump allocator on workspace ----
  size_t off = 0;
  auto alloc = [&](size_t bytes) -> void* {
    void* p = (char*)d_ws + off;
    off = (off + bytes + 255) & ~(size_t)255;
    return p;
  };
  unsigned short* xb   = (unsigned short*)alloc((size_t)N * CIN * 2);
  // transposed bf16 weights (COUT x K row-major)
  unsigned short* w1t  = (unsigned short*)alloc((size_t)CIN * COUT * 2);
  unsigned short* wst  = (unsigned short*)alloc((size_t)CIN * COUT * 2);
  unsigned short* w2t  = (unsigned short*)alloc((size_t)COUT * COUT * 2);
  unsigned short* wqt  = (unsigned short*)alloc((size_t)COUT * COUT * 2);
  unsigned short* wkt  = (unsigned short*)alloc((size_t)COUT * COUT * 2);
  unsigned short* wvt  = (unsigned short*)alloc((size_t)COUT * COUT * 2);
  float* deg   = (float*)alloc((size_t)N * 4);
  float* dis   = (float*)alloc((size_t)N * 4);
  float* h     = (float*)alloc((size_t)N * COUT * 4);
  float* ident = (float*)alloc((size_t)N * COUT * 4);
  float* agg   = (float*)alloc((size_t)N * COUT * 4);
  float* out1  = (float*)alloc((size_t)N * COUT * 4);
  float* out2  = (float*)alloc((size_t)N * COUT * 4);
  float* qtmp  = (float*)alloc((size_t)N * COUT * 4);
  unsigned short* o1b = (unsigned short*)alloc((size_t)N * COUT * 2);
  unsigned short* o2b = (unsigned short*)alloc((size_t)N * COUT * 2);
  unsigned short* qb  = (unsigned short*)alloc((size_t)N * COUT * 2);
  unsigned short* kb  = (unsigned short*)alloc((size_t)N * COUT * 2);
  // Vt padded by 32 elements: tail chunk of the contraction dim may over-read
  unsigned short* vt  = (unsigned short*)alloc(((size_t)N * COUT + 32) * 2);
  (void)ws_size; (void)n_in; (void)out_size;

  const int T = 256;
  auto blk = [](int n, int t) { return (n + t - 1) / t; };
  const int gemmBlocks = blk(N, 128);

  // convert inputs to bf16 (weights transposed for contiguous B-fragments)
  k_cvt_bf16<<<blk(N * CIN, T), T, 0, stream>>>(x, xb, N * CIN);
  k_cvt_bf16_T<<<blk(CIN * COUT, T), T, 0, stream>>>(W1, w1t, CIN, COUT);
  k_cvt_bf16_T<<<blk(CIN * COUT, T), T, 0, stream>>>(Ws, wst, CIN, COUT);
  k_cvt_bf16_T<<<blk(COUT * COUT, T), T, 0, stream>>>(W2, w2t, COUT, COUT);
  k_cvt_bf16_T<<<blk(COUT * COUT, T), T, 0, stream>>>(Wq, wqt, COUT, COUT);
  k_cvt_bf16_T<<<blk(COUT * COUT, T), T, 0, stream>>>(Wk, wkt, COUT, COUT);
  k_cvt_bf16_T<<<blk(COUT * COUT, T), T, 0, stream>>>(Wv, wvt, COUT, COUT);
  k_zero_us<<<1, 64, 0, stream>>>(vt + (size_t)N * COUT, 32);  // pad = 0

  // degrees / normalization
  k_deg_init<<<blk(N, T), T, 0, stream>>>(deg, N);
  k_deg_count<<<blk(E, T), T, 0, stream>>>(dst, deg, E);
  k_dis<<<blk(N, T), T, 0, stream>>>(deg, dis, N);

  // identity = x @ Ws + bs ;  h = x @ W1
  k_gemm_bf16<<<gemmBlocks, T, 0, stream>>>(xb, wst, bs, ident, N, CIN);
  k_gemm_bf16<<<gemmBlocks, T, 0, stream>>>(xb, w1t, nullptr, h, N, CIN);

  // GCN conv 1 aggregation + LN + SiLU
  k_selfloop<<<blk(N * COUT, T), T, 0, stream>>>(h, dis, agg, N);
  k_scatter<<<blk(E * 8, T), T, 0, stream>>>(src, dst, h, dis, agg, E);
  k_bias_ln<<<blk(N, 8), T, 0, stream>>>(agg, b1, ln1g, ln1b, out1, N, 1);
  k_cvt_bf16<<<blk(N * COUT, T), T, 0, stream>>>(out1, o1b, N * COUT);

  // GCN conv 2 + LN
  k_gemm_bf16<<<gemmBlocks, T, 0, stream>>>(o1b, w2t, nullptr, h, N, COUT);
  k_selfloop<<<blk(N * COUT, T), T, 0, stream>>>(h, dis, agg, N);
  k_scatter<<<blk(E * 8, T), T, 0, stream>>>(src, dst, h, dis, agg, E);
  k_bias_ln<<<blk(N, 8), T, 0, stream>>>(agg, b2, ln2g, ln2b, out2, N, 0);
  k_cvt_bf16<<<blk(N * COUT, T), T, 0, stream>>>(out2, o2b, N * COUT);

  // QKV projections (bf16 WMMA, f32 out, bias fused)
  k_gemm_bf16<<<gemmBlocks, T, 0, stream>>>(o2b, wqt, bq, qtmp, N, COUT);
  k_cvt_bf16<<<blk(N * COUT, T), T, 0, stream>>>(qtmp, qb, N * COUT);
  k_gemm_bf16<<<gemmBlocks, T, 0, stream>>>(o2b, wkt, bk, qtmp, N, COUT);
  k_cvt_bf16<<<blk(N * COUT, T), T, 0, stream>>>(qtmp, kb, N * COUT);
  k_gemm_bf16<<<gemmBlocks, T, 0, stream>>>(o2b, wvt, bv, qtmp, N, COUT);
  k_cvt_bf16_T<<<blk(N * COUT, T), T, 0, stream>>>(qtmp, vt, N, COUT);  // V^T

  // flash attention + residual + shortcut + SiLU -> d_out
  k_flash<<<gemmBlocks, T, 0, stream>>>(qb, kb, vt, out2, ident, out, N);
}
